// FunAudioChatAudioAttention_44581760532548
// MI455X (gfx1250) — matrix operand order repaired and checked by reading, not compiled, not run on hardware
//
#include <hip/hip_runtime.h>

// ---------------------------------------------------------------------------
// FunAudioChat attention for MI455X (gfx1250, wave32, WMMA bf16 / fp32 accum)
//   SEQ=4096, D=1024, H=16, HD=64, SCALE = 1/8
//
// Pipeline:
//   0) convert X, wq, wk, wv, wo  f32 -> bf16  (once)
//   1) Q/K/V projections: templated 64x32-tile GEMM, both operands
//      double-buffered across the k-loop (V stored transposed)
//   2) flash attention, online softmax fp32, P relayout via LDS (bf16)
//   3) output projection: same GEMM template, fp32 output + bias
// ---------------------------------------------------------------------------

#define SEQC 4096
#define DC   1024
#define HC   16
#define HDC  64
#define SCALEC 0.125f

typedef __attribute__((ext_vector_type(16))) __bf16 v16bf;
typedef __attribute__((ext_vector_type(8)))  __bf16 v8bf;
typedef __attribute__((ext_vector_type(4)))  __bf16 v4bf;
typedef __attribute__((ext_vector_type(8)))  float  v8f;

// ds_swizzle_b32 XOR mode: offset = {xor_mask[14:10], or=0, and_mask=0x1f}
#define SWZ_XOR(v, m) \
  __int_as_float(__builtin_amdgcn_ds_swizzle(__float_as_int(v), (((m) << 10) | 0x1f)))

__device__ __forceinline__ v8f wmma_bf16(v16bf a, v16bf b, v8f c) {
  return __builtin_amdgcn_wmma_f32_16x16x32_bf16(false, a, false, b, (short)0, c,
                                                 false, false);
}

__device__ __forceinline__ v16bf cat8(v8bf lo, v8bf hi) {
  return __builtin_shufflevector(lo, hi, 0, 1, 2, 3, 4, 5, 6, 7,
                                 8, 9, 10, 11, 12, 13, 14, 15);
}

// A-operand 16x32 bf16 (ISA 7.12.2): lane&15 = row M; lanes 0-15 hold
// K = 0..7 (v0-3) and 16..23 (v4-7); lanes 16-31 hold K = 8..15 and 24..31.
__device__ __forceinline__ v16bf ldA(const __bf16* __restrict__ p, int ld,
                                     int m0, int k0, int lane) {
  const __bf16* r = p + (size_t)(m0 + (lane & 15)) * ld + k0 + ((lane >> 4) << 3);
  return cat8(*(const v8bf*)r, *(const v8bf*)(r + 16));
}

// B-operand 32x16 bf16: lane&15 = column N; lanes 0-15 hold K=0..7 and 16..23,
// lanes 16-31 hold K=8..15 and 24..31.  Source src[n][k] contiguous in k.
__device__ __forceinline__ v16bf ldB(const __bf16* __restrict__ p, int ld,
                                     int n0, int k0, int lane) {
  const __bf16* r = p + (size_t)(n0 + (lane & 15)) * ld + k0 + ((lane >> 4) << 3);
  return cat8(*(const v8bf*)r, *(const v8bf*)(r + 16));
}

// ---------------------------------------------------------------------------
// Kernel 0: elementwise f32 -> bf16 (4 elements / thread, b128 in, b64 out)
// ---------------------------------------------------------------------------
__global__ void __launch_bounds__(256)
cvt_bf16_kernel(const float* __restrict__ s, __bf16* __restrict__ d, int n4) {
  const int i = blockIdx.x * 256 + threadIdx.x;
  if (i < n4) {
    const float4 v = ((const float4*)s)[i];
    v4bf o;
    o[0] = (__bf16)v.x; o[1] = (__bf16)v.y; o[2] = (__bf16)v.z; o[3] = (__bf16)v.w;
    ((v4bf*)d)[i] = o;
  }
}

// ---------------------------------------------------------------------------
// Templated GEMM:  Y[m][n] = sum_k A[m][k] * W[n][k]  (+ bias[n])
// Each wave owns a 64x32 output tile: 4 M-subtiles x 2 N-subtiles.
// Both A (4 frags) and B (2 frags) are double-buffered across the k-loop so
// the k+1 loads are in flight while the k WMMAs execute.
//   OUTMODE 0: bf16 row-major [.][DC]       (Q, K)
//   OUTMODE 1: bf16 transposed [DC][SEQC]   (V -> Vt)
//   OUTMODE 2: f32  row-major [.][DC]       (final output)
// ---------------------------------------------------------------------------
template <int OUTMODE, bool BIAS>
__global__ void __launch_bounds__(256, 1)
gemm_kernel(const __bf16* __restrict__ A, const __bf16* __restrict__ W,
            const float* __restrict__ bias, void* __restrict__ out) {
  const int lane = threadIdx.x & 31;
  const int wave = threadIdx.x >> 5;
  const int tile = blockIdx.x * 8 + wave;            // 2048 tiles of 64x32
  const int m0 = (tile / (DC / 32)) * 64;
  const int n0 = (tile % (DC / 32)) * 32;

  v8f acc[4][2] = {};
  v16bf a[4], b[2];
#pragma unroll
  for (int mi = 0; mi < 4; ++mi) a[mi] = ldA(A, DC, m0 + 16 * mi, 0, lane);
  b[0] = ldB(W, DC, n0, 0, lane);
  b[1] = ldB(W, DC, n0 + 16, 0, lane);

  for (int k0 = 0; k0 < DC; k0 += 32) {
    const int kn = (k0 + 32 < DC) ? k0 + 32 : 0;
    v16bf na[4], nb[2];
#pragma unroll
    for (int mi = 0; mi < 4; ++mi) na[mi] = ldA(A, DC, m0 + 16 * mi, kn, lane);
    nb[0] = ldB(W, DC, n0, kn, lane);
    nb[1] = ldB(W, DC, n0 + 16, kn, lane);
    // distance-2 prefetch of the A stream (global_prefetch_b8)
    __builtin_prefetch(A + (size_t)(m0 + (lane & 15)) * DC + ((k0 + 64) & (DC - 1)), 0, 1);

#pragma unroll
    for (int mi = 0; mi < 4; ++mi) {
      acc[mi][0] = wmma_bf16(a[mi], b[0], acc[mi][0]);
      acc[mi][1] = wmma_bf16(a[mi], b[1], acc[mi][1]);
    }
#pragma unroll
    for (int mi = 0; mi < 4; ++mi) a[mi] = na[mi];
    b[0] = nb[0];
    b[1] = nb[1];
  }

  // C/D layout: VGPR j -> row (half*8 + j), lane&15 -> column.
  const int cn   = lane & 15;
  const int mrow = (lane >> 4) << 3;
#pragma unroll
  for (int ni = 0; ni < 2; ++ni) {
    const int n = n0 + ni * 16 + cn;
    const float bs = BIAS ? bias[n] : 0.0f;
#pragma unroll
    for (int mi = 0; mi < 4; ++mi) {
#pragma unroll
      for (int j = 0; j < 8; ++j) {
        const int m = m0 + mi * 16 + mrow + j;
        const float v = acc[mi][ni][j] + bs;
        if (OUTMODE == 0) {
          ((__bf16*)out)[(size_t)m * DC + n] = (__bf16)v;
        } else if (OUTMODE == 1) {
          ((__bf16*)out)[(size_t)n * SEQC + m] = (__bf16)v;
        } else {
          ((float*)out)[(size_t)m * DC + n] = v;
        }
      }
    }
  }
}

// ---------------------------------------------------------------------------
// Flash attention.  One wave per (head, 16-query block).  Streams keys/values
// in tiles of 32, online softmax in fp32, P*V via bf16 WMMA.  Current-tile V
// loads and next-tile K loads are issued right after the scores WMMA so they
// overlap the softmax VALU work.  Row statistics are reduced across the
// 16-lane halves with ds_swizzle XOR butterflies (C-layout: a row's 16
// columns live in 16 lanes of one VGPR).
// ---------------------------------------------------------------------------
__global__ void __launch_bounds__(128, 1)
attn_fa_kernel(const __bf16* __restrict__ Qb, const __bf16* __restrict__ Kb,
               const __bf16* __restrict__ Vt, __bf16* __restrict__ Ob) {
  __shared__ __bf16 pbuf[4][16 * 32];                // per-wave P relayout buf
  const int lane = threadIdx.x & 31;
  const int wave = threadIdx.x >> 5;
  const int q0 = (blockIdx.x * 4 + wave) * 16;       // query block
  const int h  = blockIdx.y;
  const int kq = h * HDC;                            // head column base

  // Q fragments, hd=64 split into two K=32 chunks (resident all loop long)
  const v16bf aq0 = ldA(Qb, DC, q0, kq,      lane);
  const v16bf aq1 = ldA(Qb, DC, q0, kq + 32, lane);

  v8f o0 = {}, o1 = {}, o2 = {}, o3 = {};            // O accum, N = hd 0..63
  float m_run[8], l_run[8];
#pragma unroll
  for (int j = 0; j < 8; ++j) { m_run[j] = -3.0e38f; l_run[j] = 0.0f; }

  __bf16* buf = pbuf[wave];
  const int cn   = lane & 15;          // C-layout column
  const int mrow = (lane >> 4) << 3;   // C-layout row base
  const int am   = lane & 15;          // A-layout row
  const int akhi = (lane >> 4) << 3;   // A-layout K base

  // preload K fragments for the first key tile
  v16bf bk0 = ldB(Kb, DC, 0,  kq,      lane);
  v16bf bk1 = ldB(Kb, DC, 0,  kq + 32, lane);
  v16bf bk2 = ldB(Kb, DC, 16, kq,      lane);
  v16bf bk3 = ldB(Kb, DC, 16, kq + 32, lane);

  for (int kt = 0; kt < SEQC; kt += 32) {
    // S = Q * K^T for 32 keys: two 16x16 C tiles, K-accumulated over hd=64
    v8f s0 = {}, s1 = {};
    s0 = wmma_bf16(aq0, bk0, s0);
    s0 = wmma_bf16(aq1, bk1, s0);
    s1 = wmma_bf16(aq0, bk2, s1);
    s1 = wmma_bf16(aq1, bk3, s1);

    // issue current-tile V loads + next-tile K loads; they fly during softmax
    v16bf bv0 = ldB(Vt, SEQC, kq,      kt, lane);
    v16bf bv1 = ldB(Vt, SEQC, kq + 16, kt, lane);
    v16bf bv2 = ldB(Vt, SEQC, kq + 32, kt, lane);
    v16bf bv3 = ldB(Vt, SEQC, kq + 48, kt, lane);
    const int ktn = (kt + 32 < SEQC) ? kt + 32 : 0;
    bk0 = ldB(Kb, DC, ktn,      kq,      lane);
    bk1 = ldB(Kb, DC, ktn,      kq + 32, lane);
    bk2 = ldB(Kb, DC, ktn + 16, kq,      lane);
    bk3 = ldB(Kb, DC, ktn + 16, kq + 32, lane);

    // online softmax over this 16x32 tile (fp32)
#pragma unroll
    for (int j = 0; j < 8; ++j) {
      const float slo = s0[j] * SCALEC;
      const float shi = s1[j] * SCALEC;
      float tm = fmaxf(slo, shi);
      tm = fmaxf(tm, SWZ_XOR(tm, 1));
      tm = fmaxf(tm, SWZ_XOR(tm, 2));
      tm = fmaxf(tm, SWZ_XOR(tm, 4));
      tm = fmaxf(tm, SWZ_XOR(tm, 8));
      const float mn   = fmaxf(m_run[j], tm);
      const float corr = __expf(m_run[j] - mn);
      m_run[j] = mn;
      const float plo = __expf(slo - mn);
      const float phi = __expf(shi - mn);
      float rs = plo + phi;
      rs += SWZ_XOR(rs, 1);
      rs += SWZ_XOR(rs, 2);
      rs += SWZ_XOR(rs, 4);
      rs += SWZ_XOR(rs, 8);
      l_run[j] = l_run[j] * corr + rs;
      o0[j] *= corr; o1[j] *= corr; o2[j] *= corr; o3[j] *= corr;
      // stage P (C-layout) into LDS row-major [m][k], bf16
      buf[(mrow + j) * 32 + cn]      = (__bf16)plo;
      buf[(mrow + j) * 32 + 16 + cn] = (__bf16)phi;
    }
    __syncthreads();
    // reload P in WMMA A-layout: two ds_load_b128 per lane
    const __bf16* rp = &buf[am * 32 + akhi];
    const v16bf ap = cat8(*(const v8bf*)rp, *(const v8bf*)(rp + 16));
    __syncthreads();

    // O += P * V  (V^T rows are contiguous in keys)
    o0 = wmma_bf16(ap, bv0, o0);
    o1 = wmma_bf16(ap, bv1, o1);
    o2 = wmma_bf16(ap, bv2, o2);
    o3 = wmma_bf16(ap, bv3, o3);
  }

  // normalize rows and store bf16 into the [S][D] attention-output buffer
#pragma unroll
  for (int j = 0; j < 8; ++j) {
    const float inv = 1.0f / l_run[j];
    const size_t row = (size_t)(q0 + mrow + j) * DC + kq;
    Ob[row + cn]      = (__bf16)(o0[j] * inv);
    Ob[row + 16 + cn] = (__bf16)(o1[j] * inv);
    Ob[row + 32 + cn] = (__bf16)(o2[j] * inv);
    Ob[row + 48 + cn] = (__bf16)(o3[j] * inv);
  }
}

// ---------------------------------------------------------------------------
extern "C" void kernel_launch(void* const* d_in, const int* in_sizes, int n_in,
                              void* d_out, int out_size, void* d_ws, size_t ws_size,
                              hipStream_t stream) {
  (void)in_sizes; (void)n_in; (void)out_size; (void)ws_size;
  const float* X  = (const float*)d_in[0];
  const float* wq = (const float*)d_in[1];
  const float* bq = (const float*)d_in[2];
  const float* wk = (const float*)d_in[3];
  const float* wv = (const float*)d_in[4];
  const float* bv = (const float*)d_in[5];
  const float* wo = (const float*)d_in[6];
  const float* bo = (const float*)d_in[7];
  float* Y = (float*)d_out;

  // workspace layout (bf16 elements):
  //   Xb[S*D] | Wqb | Wkb | Wvb | Wob (D*D each) | Qb | Kb | Vt | Ob (S*D each)
  const size_t SD = (size_t)SEQC * DC;
  const size_t DD = (size_t)DC * DC;
  __bf16* Xb  = (__bf16*)d_ws;
  __bf16* Wqb = Xb + SD;
  __bf16* Wkb = Wqb + DD;
  __bf16* Wvb = Wkb + DD;
  __bf16* Wob = Wvb + DD;
  __bf16* Qb  = Wob + DD;
  __bf16* Kb  = Qb + SD;
  __bf16* Vt  = Kb + SD;
  __bf16* Ob  = Vt + SD;

  // 0) one-time f32 -> bf16 conversions (memory bound, ~20 MB total)
  cvt_bf16_kernel<<<dim3((int)(SD / 4 / 256)), dim3(256), 0, stream>>>(X,  Xb,  (int)(SD / 4));
  cvt_bf16_kernel<<<dim3((int)(DD / 4 / 256)), dim3(256), 0, stream>>>(wq, Wqb, (int)(DD / 4));
  cvt_bf16_kernel<<<dim3((int)(DD / 4 / 256)), dim3(256), 0, stream>>>(wk, Wkb, (int)(DD / 4));
  cvt_bf16_kernel<<<dim3((int)(DD / 4 / 256)), dim3(256), 0, stream>>>(wv, Wvb, (int)(DD / 4));
  cvt_bf16_kernel<<<dim3((int)(DD / 4 / 256)), dim3(256), 0, stream>>>(wo, Wob, (int)(DD / 4));

  // 64x32 tiles, 8 waves per block
  const int gemm_blocks = (SEQC / 64) * (DC / 32) / 8;   // 256 blocks

  // 1) projections
  gemm_kernel<0, true ><<<dim3(gemm_blocks), dim3(256), 0, stream>>>(Xb, Wqb, bq, Qb);
  gemm_kernel<0, false><<<dim3(gemm_blocks), dim3(256), 0, stream>>>(Xb, Wkb, nullptr, Kb);
  gemm_kernel<1, true ><<<dim3(gemm_blocks), dim3(256), 0, stream>>>(Xb, Wvb, bv, Vt);

  // 2) attention
  attn_fa_kernel<<<dim3(SEQC / 64, HC), dim3(128), 0, stream>>>(Qb, Kb, Vt, Ob);

  // 3) output projection (fp32 result)
  gemm_kernel<2, true ><<<dim3(gemm_blocks), dim3(256), 0, stream>>>(Ob, Wob, bo, Y);
}
